// MoE_49031346651834
// MI455X (gfx1250) — matrix operand order repaired and checked by reading, not compiled, not run on hardware
//
#include <hip/hip_runtime.h>
#include <hip/hip_bf16.h>
#include <cstdint>

typedef __attribute__((ext_vector_type(16))) _Float16 v16h;
typedef __attribute__((ext_vector_type(8)))  _Float16 v8h;
typedef __attribute__((ext_vector_type(8)))  float    v8f;

constexpr int BATCH = 256;   // B
constexpr int DIM   = 512;   // D
constexpr int NEXP  = 8;     // K experts
constexpr int GDIM  = 128;   // gate hidden
constexpr int PDIM  = 64;    // phase dim
constexpr int KDIM  = NEXP * DIM;     // 4096 fused reduction dim

constexpr int SSPLIT = 8;             // K-dim split factor (occupancy)
constexpr int KRANGE = KDIM / SSPLIT; // 512 K per block
constexpr int KC     = 128;           // K chunk staged in LDS per iteration
constexpr int NCHUNK = KRANGE / KC;   // 4 pipelined chunks
constexpr int NCOLS  = 32;            // output cols per block (2 N-tiles/wave)
constexpr int LSTRIDE = KC + 8;       // padded halves per col (272B: 4-bank skew)

__device__ __forceinline__ float elu1(float x) {
    return x > 0.0f ? x : (__expf(x) - 1.0f);
}

// ---------------------------------------------------------------------------
// Gate: phase[256,64] -> ELU(Linear) -> ELU(Linear) -> Linear -> softmax(8)
// ---------------------------------------------------------------------------
__global__ void gate_kernel(const float* __restrict__ phase,
                            const float* __restrict__ gW1, const float* __restrict__ gb1,
                            const float* __restrict__ gW2, const float* __restrict__ gb2,
                            const float* __restrict__ gW3, const float* __restrict__ gb3,
                            float* __restrict__ coef)
{
    __shared__ float sph[PDIM];
    __shared__ float s1[GDIM];
    __shared__ float s2[GDIM];
    __shared__ float s3[NEXP];
    const int b = blockIdx.x;
    const int t = threadIdx.x; // 0..127
    if (t < PDIM) sph[t] = phase[b * PDIM + t];
    __syncthreads();

    float acc = gb1[t];
    #pragma unroll 4
    for (int p = 0; p < PDIM; ++p) acc = fmaf(sph[p], gW1[p * GDIM + t], acc);
    s1[t] = elu1(acc);
    __syncthreads();

    acc = gb2[t];
    #pragma unroll 4
    for (int g = 0; g < GDIM; ++g) acc = fmaf(s1[g], gW2[g * GDIM + t], acc);
    s2[t] = elu1(acc);
    __syncthreads();

    if (t < NEXP) {
        acc = gb3[t];
        #pragma unroll 4
        for (int g = 0; g < GDIM; ++g) acc = fmaf(s2[g], gW3[g * NEXP + t], acc);
        s3[t] = acc;
    }
    __syncthreads();

    if (t == 0) {
        float m = s3[0];
        for (int k = 1; k < NEXP; ++k) m = fmaxf(m, s3[k]);
        float e[NEXP]; float sum = 0.0f;
        for (int k = 0; k < NEXP; ++k) { e[k] = __expf(s3[k] - m); sum += e[k]; }
        float inv = 1.0f / sum;
        for (int k = 0; k < NEXP; ++k) coef[b * NEXP + k] = e[k] * inv;
    }
}

// ---------------------------------------------------------------------------
// Split fp32 weights w[kd][o] into f16 hi/lo, stored transposed as [o][kd]
// so WMMA B-fragments are contiguous. LDS-tiled 32x32 transpose.
// ---------------------------------------------------------------------------
__global__ void split_w_kernel(const float* __restrict__ w,    // [KDIM][DIM]
                               _Float16* __restrict__ whiT,    // [DIM][KDIM]
                               _Float16* __restrict__ wloT)
{
    __shared__ float tile[32][33];
    const int kd0 = blockIdx.x * 32;
    const int o0  = blockIdx.y * 32;
    const int tx = threadIdx.x; // 0..31
    const int ty = threadIdx.y; // 0..7
    #pragma unroll
    for (int j = 0; j < 4; ++j) {
        int kd = kd0 + ty + 8 * j;
        tile[ty + 8 * j][tx] = w[(size_t)kd * DIM + o0 + tx];
    }
    __syncthreads();
    #pragma unroll
    for (int j = 0; j < 4; ++j) {
        int o = o0 + ty + 8 * j;
        float v = tile[tx][ty + 8 * j];
        _Float16 hi = (_Float16)v;
        float lo = v - (float)hi;
        whiT[(size_t)o * KDIM + kd0 + tx] = hi;
        wloT[(size_t)o * KDIM + kd0 + tx] = (_Float16)lo;
    }
}

// ---------------------------------------------------------------------------
// Fused A operand: t[b, k*D+i] = coef[b,k] * h[b,i], split into f16 hi/lo.
// ---------------------------------------------------------------------------
__global__ void prep_kernel(const float* __restrict__ h,     // [BATCH][DIM]
                            const float* __restrict__ coef,  // [BATCH][NEXP]
                            _Float16* __restrict__ thi,      // [BATCH][KDIM]
                            _Float16* __restrict__ tlo)
{
    int idx = blockIdx.x * blockDim.x + threadIdx.x;
    int b  = idx >> 12;
    int kd = idx & (KDIM - 1);
    float v = coef[b * NEXP + (kd >> 9)] * h[b * DIM + (kd & (DIM - 1))];
    _Float16 hi = (_Float16)v;
    thi[idx] = hi;
    tlo[idx] = (_Float16)(v - (float)hi);
}

// ---------------------------------------------------------------------------
// K-split GEMM block: 16 waves (512 thr) = 16 M-tiles x 32 cols, one K-range
// of 512. B (hi+lo) double-buffered in LDS via CDNA5 async global->LDS DMA:
// chunk i+1 streams into one buffer (ASYNCcnt-tracked) while chunk i's WMMA
// burst consumes the other via ds_load_b128. Async loads complete in order,
// so `s_wait_asynccnt 2` releases exactly the previous chunk's 2 loads.
// 3x f16 WMMA per K-step reproduces ~fp32. Writes fp32 partials (no atomics:
// bitwise deterministic).
// ---------------------------------------------------------------------------
__global__ __launch_bounds__(512)
void moe_gemm_kernel(const _Float16* __restrict__ thi,
                     const _Float16* __restrict__ tlo,
                     const _Float16* __restrict__ whiT,   // [DIM][KDIM]
                     const _Float16* __restrict__ wloT,
                     float* __restrict__ partial)          // [SSPLIT][BATCH][DIM]
{
    __shared__ __attribute__((aligned(16))) _Float16 sB[2][2][NCOLS * LSTRIDE];

    const int cg   = blockIdx.x & 15;     // 16 col-groups of 32 cols
    const int s    = blockIdx.x >> 4;     // 0..SSPLIT-1
    const int col0 = cg * NCOLS;
    const int kbase = s * KRANGE;

    const int tid  = threadIdx.x;
    const int wave = tid >> 5;            // = M tile (0..15)
    const int lane = tid & 31;
    const int hsel = lane >> 4;           // 0: lanes 0-15, 1: lanes 16-31
    const int mrow = wave * 16 + (lane & 15);

    // A layout (16-bit A 16x32): lanes 0-15 hold K 0-7 & 16-23; lanes 16-31 +8.
    const _Float16* arow_hi = thi + (size_t)mrow * KDIM + kbase + hsel * 8;
    const _Float16* arow_lo = tlo + (size_t)mrow * KDIM + kbase + hsel * 8;

    // This thread's two 16B copy units (fixed across chunks): arr/col/sub.
    const int u0arr = tid >> 8 & 0;       // thread 0..511 -> first unit is hi half
    (void)u0arr;

    auto issue_chunk = [&](int chunk, int buf) {
        #pragma unroll
        for (int uu = 0; uu < 2; ++uu) {
            int u   = tid + uu * 512;     // 1024 units total
            int arr = u >> 9;             // 0 = hi, 1 = lo
            int rem = u & 511;
            int col = rem >> 4;           // 0..31
            int sub = rem & 15;           // 16B unit within the col's K chunk
            const _Float16* gsrc = (arr ? wloT : whiT)
                + (size_t)(col0 + col) * KDIM + kbase + chunk + sub * 8;
            // low 32 bits of an LDS pointer == LDS byte address (aperture rule)
            uint32_t ldsoff =
                (uint32_t)(uintptr_t)(&sB[buf][arr][col * LSTRIDE + sub * 8]);
            asm volatile("global_load_async_to_lds_b128 %0, %1, off"
                         :: "v"(ldsoff),
                            "v"((unsigned long long)(uintptr_t)gsrc)
                         : "memory");
        }
    };

    v8f accv[2] = {{}, {}};

    issue_chunk(0, 0);

    #pragma unroll
    for (int ci = 0; ci < NCHUNK; ++ci) {
        const int chunk = ci * KC;
        if (ci + 1 < NCHUNK) {
            // stream next chunk into the other buffer, then wait only for the
            // previous chunk's 2 (in-order) async loads.
            issue_chunk(chunk + KC, (ci + 1) & 1);
            asm volatile("s_wait_asynccnt 0x2" ::: "memory");
        } else {
            asm volatile("s_wait_asynccnt 0x0" ::: "memory");
        }
        __syncthreads();   // chunk ci resident in sB[ci&1] for all waves

        const _Float16* sHi = &sB[ci & 1][0][0];
        const _Float16* sLo = &sB[ci & 1][1][0];

        #pragma unroll
        for (int kk = 0; kk < KC; kk += 32) {
            union { v16h v; v8h h[2]; } ahi, alo, bh, bl;
            const _Float16* ah = arow_hi + chunk + kk;
            const _Float16* al = arow_lo + chunk + kk;
            ahi.h[0] = *(const v8h*)(ah);
            ahi.h[1] = *(const v8h*)(ah + 16);
            alo.h[0] = *(const v8h*)(al);
            alo.h[1] = *(const v8h*)(al + 16);
            #pragma unroll
            for (int j = 0; j < 2; ++j) {
                // B layout (32x16): lanes 0-15 -> K 0-7 & 16-23, lanes 16-31 +8
                const _Float16* bp_h =
                    sHi + (j * 16 + (lane & 15)) * LSTRIDE + kk + hsel * 8;
                const _Float16* bp_l =
                    sLo + (j * 16 + (lane & 15)) * LSTRIDE + kk + hsel * 8;
                bh.h[0] = *(const v8h*)(bp_h);
                bh.h[1] = *(const v8h*)(bp_h + 16);
                bl.h[0] = *(const v8h*)(bp_l);
                bl.h[1] = *(const v8h*)(bp_l + 16);
                accv[j] = __builtin_amdgcn_wmma_f32_16x16x32_f16(
                    false, ahi.v, false, bh.v, (short)0, accv[j], false, false);
                accv[j] = __builtin_amdgcn_wmma_f32_16x16x32_f16(
                    false, ahi.v, false, bl.v, (short)0, accv[j], false, false);
                accv[j] = __builtin_amdgcn_wmma_f32_16x16x32_f16(
                    false, alo.v, false, bh.v, (short)0, accv[j], false, false);
            }
        }
        __syncthreads();   // all waves done with sB[ci&1] before it is reused
    }

    // C/D layout: VGPR r, lanes 0-15 -> M=r, lanes 16-31 -> M=8+r; N = lane%16.
    const int mhalf = hsel * 8;
    #pragma unroll
    for (int j = 0; j < 2; ++j) {
        int o = col0 + j * 16 + (lane & 15);
        #pragma unroll
        for (int r = 0; r < 8; ++r) {
            int brow = wave * 16 + r + mhalf;
            partial[((size_t)s * BATCH + brow) * DIM + o] = accv[j][r];
        }
    }
}

// ---------------------------------------------------------------------------
// Deterministic K-split reduction + mixed bias + optional ELU/dropout.
// MODE 0: ELU + dropout(u, p=0.25)   MODE 1: plain (final layer)
// ---------------------------------------------------------------------------
template<int MODE>
__global__ void epilogue_kernel(const float* __restrict__ partial, // [SSPLIT][B*D]
                                const float* __restrict__ coef,    // [B][NEXP]
                                const float* __restrict__ bias,    // [NEXP][DIM]
                                const float* __restrict__ u,       // [B][DIM] / null
                                float* __restrict__ out)           // [B][DIM]
{
    int idx = blockIdx.x * blockDim.x + threadIdx.x;  // over BATCH*DIM
    int b = idx >> 9;
    int o = idx & (DIM - 1);
    float v = 0.0f;
    #pragma unroll
    for (int sp = 0; sp < SSPLIT; ++sp)
        v += partial[(size_t)sp * BATCH * DIM + idx];
    #pragma unroll
    for (int k = 0; k < NEXP; ++k)
        v = fmaf(coef[b * NEXP + k], bias[k * DIM + o], v);
    if (MODE == 0) {
        v = elu1(v);
        v *= (u[idx] >= 0.25f) ? (1.0f / 0.75f) : 0.0f;
    }
    out[idx] = v;
}

// ---------------------------------------------------------------------------
extern "C" void kernel_launch(void* const* d_in, const int* in_sizes, int n_in,
                              void* d_out, int out_size, void* d_ws, size_t ws_size,
                              hipStream_t stream)
{
    (void)in_sizes; (void)n_in; (void)out_size; (void)ws_size;
    const float* x     = (const float*)d_in[0];
    const float* phase = (const float*)d_in[1];
    const float* gW1   = (const float*)d_in[2];
    const float* gb1   = (const float*)d_in[3];
    const float* gW2   = (const float*)d_in[4];
    const float* gb2   = (const float*)d_in[5];
    const float* gW3   = (const float*)d_in[6];
    const float* gb3   = (const float*)d_in[7];
    const float* w0    = (const float*)d_in[8];
    const float* b0    = (const float*)d_in[9];
    const float* w1    = (const float*)d_in[10];
    const float* b1    = (const float*)d_in[11];
    const float* w2    = (const float*)d_in[12];
    const float* b2    = (const float*)d_in[13];
    const float* u1    = (const float*)d_in[14];
    const float* u2    = (const float*)d_in[15];
    float* out = (float*)d_out;

    char* ws = (char*)d_ws;
    size_t off = 0;
    auto carve = [&](size_t bytes) -> char* {
        char* p = ws + off;
        off += (bytes + 255) & ~(size_t)255;
        return p;
    };
    float*    coef    = (float*)   carve((size_t)BATCH * NEXP * sizeof(float));
    float*    h1      = (float*)   carve((size_t)BATCH * DIM  * sizeof(float));
    float*    h2      = (float*)   carve((size_t)BATCH * DIM  * sizeof(float));
    float*    partial = (float*)   carve((size_t)SSPLIT * BATCH * DIM * sizeof(float));
    _Float16* thi     = (_Float16*)carve((size_t)BATCH * KDIM * 2);
    _Float16* tlo     = (_Float16*)carve((size_t)BATCH * KDIM * 2);
    _Float16* whiT[3]; _Float16* wloT[3];
    for (int l = 0; l < 3; ++l) {
        whiT[l] = (_Float16*)carve((size_t)DIM * KDIM * 2);
        wloT[l] = (_Float16*)carve((size_t)DIM * KDIM * 2);
    }

    // 1) gate coefficients
    gate_kernel<<<BATCH, GDIM, 0, stream>>>(phase, gW1, gb1, gW2, gb2, gW3, gb3, coef);

    // 2) split+transpose expert weights into f16 hi/lo (L2-resident afterwards)
    dim3 sgrid(KDIM / 32, DIM / 32), sblk(32, 8);
    split_w_kernel<<<sgrid, sblk, 0, stream>>>(w0, whiT[0], wloT[0]);
    split_w_kernel<<<sgrid, sblk, 0, stream>>>(w1, whiT[1], wloT[1]);
    split_w_kernel<<<sgrid, sblk, 0, stream>>>(w2, whiT[2], wloT[2]);

    const int prep_blocks = (BATCH * KDIM) / 256;          // 4096
    const int gemm_blocks = 16 * SSPLIT;                   // 128 blocks x 16 waves
    const int epi_blocks  = (BATCH * DIM) / 256;           // 512

    // 3) layer 0: h1 = dropout(elu(mix(x)))
    prep_kernel<<<prep_blocks, 256, 0, stream>>>(x, coef, thi, tlo);
    moe_gemm_kernel<<<gemm_blocks, 512, 0, stream>>>(thi, tlo, whiT[0], wloT[0], partial);
    epilogue_kernel<0><<<epi_blocks, 256, 0, stream>>>(partial, coef, b0, u1, h1);

    // 4) layer 1: h2 = dropout(elu(mix(h1)))
    prep_kernel<<<prep_blocks, 256, 0, stream>>>(h1, coef, thi, tlo);
    moe_gemm_kernel<<<gemm_blocks, 512, 0, stream>>>(thi, tlo, whiT[1], wloT[1], partial);
    epilogue_kernel<0><<<epi_blocks, 256, 0, stream>>>(partial, coef, b1, u2, h2);

    // 5) layer 2: out = mix(h2)   (no activation)
    prep_kernel<<<prep_blocks, 256, 0, stream>>>(h2, coef, thi, tlo);
    moe_gemm_kernel<<<gemm_blocks, 512, 0, stream>>>(thi, tlo, whiT[2], wloT[2], partial);
    epilogue_kernel<1><<<epi_blocks, 256, 0, stream>>>(partial, coef, b2, nullptr, out);
}
